// TkStateSpaceModule_57397942943972
// MI455X (gfx1250) — matrix operand order repaired and checked by reading, not compiled, not run on hardware
//
#include <hip/hip_runtime.h>
#include <hip/hip_bf16.h>

typedef __attribute__((ext_vector_type(16))) __bf16 v16bf;
typedef __attribute__((ext_vector_type(8)))  float  v8f;

#define D_INPUT  128
#define D_STATE  256
#define D_OUTPUT 128
#define BATCH    32
#define SEQ_LEN  4096
#define L_CHUNK  64
#define NCHUNK   (SEQ_LEN / L_CHUNK)   // 64 chunks
#define BT_TILES (BATCH / 16)          // 2 batch tiles of 16 rows
#define FRAG_U16 512                   // 32 lanes * 16 bf16 per fragment

union FragU { v16bf v; uint4 q[2]; unsigned short s[16]; };

__device__ __forceinline__ unsigned short f2bf(float f) {
  unsigned int u = __float_as_uint(f);
  return (unsigned short)((u + 0x7FFFu + ((u >> 16) & 1u)) >> 16);  // RNE
}

__device__ __forceinline__ v8f zero8() {
  v8f z = {0.f, 0.f, 0.f, 0.f, 0.f, 0.f, 0.f, 0.f};
  return z;
}

__device__ __forceinline__ v8f wmma_bf16(v16bf a, v16bf b, v8f c) {
  return __builtin_amdgcn_wmma_f32_16x16x32_bf16(false, a, false, b, (short)0, c,
                                                 false, false);
}

// A-fragment (16x32 MxK bf16) from a row-major bf16 buffer, leading dim `ld` elems.
// lanes 0-15: row M=lane, K 0..15; lanes 16-31: K 16..31.
// 16 contiguous elements -> two 16B loads per lane.
__device__ __forceinline__ v16bf afrag(const unsigned short* buf, int ld, int kbase,
                                       int lane) {
  const int row = lane & 15, half = lane >> 4;
  const unsigned short* p = buf + row * ld + kbase + half * 16;
  FragU f;
  f.q[0] = *(const uint4*)(p);
  f.q[1] = *(const uint4*)(p + 8);
  return f.v;
}

// B-fragment (32x16 KxN bf16) gathered from row-major W[k][n], leading dim ld.
// Slow path (strided u16 loads) -- used only in the tiny one-shot kernels.
__device__ __forceinline__ v16bf bfrag(const unsigned short* W, int ld, int kbase,
                                       int nbase, int lane) {
  const int n = lane & 15, half = lane >> 4;
  const unsigned short* p = W + (kbase + half * 16) * ld + nbase + n;
  FragU f;
#pragma unroll
  for (int e = 0; e < 16; ++e) f.s[e] = p[e * ld];
  return f.v;
}

// B-fragment from pre-swizzled fragment-linear storage: 32B per lane, coalesced.
__device__ __forceinline__ v16bf ldfrag(const unsigned short* base, int lane) {
  const unsigned short* p = base + lane * 16;
  FragU f;
  f.q[0] = *(const uint4*)(p);
  f.q[1] = *(const uint4*)(p + 8);
  return f.v;
}

// f32 16x16 C/D-layout accumulator -> bf16 into row-major LDS buffer
__device__ __forceinline__ void acc_to_lds_bf16(unsigned short* buf, int ld, int n0,
                                                int lane, v8f acc) {
  const int half = lane >> 4, col = n0 + (lane & 15);
#pragma unroll
  for (int r = 0; r < 8; ++r) buf[(half * 8 + r) * ld + col] = f2bf(acc[r]);
}

__device__ __forceinline__ void acc_to_gmem_f32(float* p, int ld, int n0, int lane,
                                                v8f acc) {
  const int half = lane >> 4, col = n0 + (lane & 15);
#pragma unroll
  for (int r = 0; r < 8; ++r) p[(half * 8 + r) * ld + col] = acc[r];
}

__device__ __forceinline__ v8f acc_from_gmem_f32(const float* p, int ld, int n0,
                                                 int lane) {
  const int half = lane >> 4, col = n0 + (lane & 15);
  v8f a;
#pragma unroll
  for (int r = 0; r < 8; ++r) a[r] = p[(half * 8 + r) * ld + col];
  return a;
}

// ---------------------------------------------------------------------------
// Setup. Raw row-major W_A = (I + dt*A)^T (for the matrix-power chain), plus
// fragment-linear ("swizzled") copies of all four weights:
//   swz[((nt*NKC + kc)*32 + lane)*16 + e]  with  n = nt*16 + (lane&15),
//   k = kc*32 + (lane>>4)*16 + e
// so the hot kernels load each B-fragment as two coalesced b128 reads.
// ---------------------------------------------------------------------------
__global__ void k_setup(const float* __restrict__ A, const float* __restrict__ B,
                        const float* __restrict__ C, const float* __restrict__ D,
                        const float* __restrict__ log_dt,
                        unsigned short* WA,
                        unsigned short* WAs, unsigned short* WBs,
                        unsigned short* WCs, unsigned short* WDs) {
  const float dt = __expf(log_dt[0]);
  const int stride = gridDim.x * blockDim.x;
  const int tid = blockIdx.x * blockDim.x + threadIdx.x;
  // raw W_A[k][n] = dt*A[n][k] + I
  for (int i = tid; i < 256 * 256; i += stride) {
    int k = i >> 8, n = i & 255;
    WA[i] = f2bf(dt * A[n * 256 + k] + (k == n ? 1.f : 0.f));
  }
  // swizzled W_A: 16 n-tiles x 8 k-tiles
  for (int i = tid; i < 16 * 8 * FRAG_U16; i += stride) {
    int e = i & 15, lane = (i >> 4) & 31, rest = i >> 9;
    int kc = rest & 7, nt = rest >> 3;
    int n = nt * 16 + (lane & 15);
    int k = kc * 32 + (lane >> 4) * 16 + e;
    WAs[i] = f2bf(dt * A[n * 256 + k] + (k == n ? 1.f : 0.f));
  }
  // swizzled W_B: 16 n-tiles x 4 k-tiles (K = 128)
  for (int i = tid; i < 16 * 4 * FRAG_U16; i += stride) {
    int e = i & 15, lane = (i >> 4) & 31, rest = i >> 9;
    int kc = rest & 3, nt = rest >> 2;
    int n = nt * 16 + (lane & 15);
    int k = kc * 32 + (lane >> 4) * 16 + e;
    WBs[i] = f2bf(dt * B[n * 128 + k]);
  }
  // swizzled W_C: 8 n-tiles x 8 k-tiles (N = 128, K = 256)
  for (int i = tid; i < 8 * 8 * FRAG_U16; i += stride) {
    int e = i & 15, lane = (i >> 4) & 31, rest = i >> 9;
    int kc = rest & 7, nt = rest >> 3;
    int n = nt * 16 + (lane & 15);
    int k = kc * 32 + (lane >> 4) * 16 + e;
    WCs[i] = f2bf(C[n * 256 + k]);
  }
  // swizzled W_D: 8 n-tiles x 4 k-tiles (N = 128, K = 128)
  for (int i = tid; i < 8 * 4 * FRAG_U16; i += stride) {
    int e = i & 15, lane = (i >> 4) & 31, rest = i >> 9;
    int kc = rest & 3, nt = rest >> 2;
    int n = nt * 16 + (lane & 15);
    int k = kc * 32 + (lane >> 4) * 16 + e;
    WDs[i] = f2bf(D[n * 128 + k]);
  }
}

// ---------------------------------------------------------------------------
// 256x256 bf16 matrix squaring (W_A^64 via 6 launches). One-shot, tiny.
// ---------------------------------------------------------------------------
__global__ __launch_bounds__(512) void k_matsq(const unsigned short* __restrict__ Win,
                                               unsigned short* __restrict__ Wout) {
  const int lane = threadIdx.x & 31, w = threadIdx.x >> 5;
  const int mb = blockIdx.x;
  const int n0 = w * 16;
  v16bf bf[8];
#pragma unroll
  for (int kc = 0; kc < 8; ++kc) bf[kc] = bfrag(Win, 256, kc * 32, n0, lane);
  v8f acc = zero8();
#pragma unroll
  for (int kc = 0; kc < 8; ++kc)
    acc = wmma_bf16(afrag(Win + mb * 16 * 256, 256, kc * 32, lane), bf[kc], acc);
  const int half = lane >> 4, col = n0 + (lane & 15);
#pragma unroll
  for (int r = 0; r < 8; ++r)
    Wout[(mb * 16 + half * 8 + r) * 256 + col] = f2bf(acc[r]);
}

// ---------------------------------------------------------------------------
// Pass 1: per (chunk, batch-tile) local scan with h0 = 0; emit end-state E_c.
// ---------------------------------------------------------------------------
__global__ __launch_bounds__(512, 1) void k_endstate(
    const float* __restrict__ x, const unsigned short* __restrict__ WAs,
    const unsigned short* __restrict__ WBs, float* __restrict__ E) {
  __shared__ unsigned short h_sh[16 * 256];
  __shared__ unsigned short x_sh[16 * 128];
  const int tid = threadIdx.x, lane = tid & 31, w = tid >> 5;
  const int chunk = blockIdx.x, bt = blockIdx.y;
  const int b0 = bt * 16, t0 = chunk * L_CHUNK;
  const int n0 = w * 16;

  v16bf wa[8], wb[4];
#pragma unroll
  for (int kc = 0; kc < 8; ++kc)
    wa[kc] = ldfrag(WAs + (w * 8 + kc) * FRAG_U16, lane);
#pragma unroll
  for (int kc = 0; kc < 4; ++kc)
    wb[kc] = ldfrag(WBs + (w * 4 + kc) * FRAG_U16, lane);

  for (int i = tid; i < 16 * 256; i += 512) h_sh[i] = 0;

  const int e0 = tid * 4, srow = e0 >> 7, scol = e0 & 127;  // 512*4 = 16*128
  const float* xbase = x + (size_t)(b0 + srow) * SEQ_LEN * D_INPUT + scol;

  v8f acc = zero8();
  __syncthreads();

  for (int t = 0; t < L_CHUNK; ++t) {
    const float* xp = xbase + (size_t)(t0 + t) * D_INPUT;
    float4 xv = *(const float4*)xp;
    if (t + 1 < L_CHUNK) __builtin_prefetch(xp + D_INPUT, 0, 0);
    uint2 pk;
    pk.x = (unsigned)f2bf(xv.x) | ((unsigned)f2bf(xv.y) << 16);
    pk.y = (unsigned)f2bf(xv.z) | ((unsigned)f2bf(xv.w) << 16);
    *(uint2*)&x_sh[srow * 128 + scol] = pk;
    __syncthreads();

    v8f a = zero8();
#pragma unroll
    for (int kc = 0; kc < 8; ++kc)
      a = wmma_bf16(afrag(h_sh, 256, kc * 32, lane), wa[kc], a);
#pragma unroll
    for (int kc = 0; kc < 4; ++kc)
      a = wmma_bf16(afrag(x_sh, 128, kc * 32, lane), wb[kc], a);
    __syncthreads();                       // all h_sh/x_sh reads done
    acc_to_lds_bf16(h_sh, 256, n0, lane, a);
    acc = a;
    __syncthreads();                       // h_t visible to all waves
  }
  acc_to_gmem_f32(E + (size_t)(chunk * BT_TILES + bt) * 16 * 256, 256, n0, lane, acc);
}

// ---------------------------------------------------------------------------
// Chunk prefix: h_init_c = h_init_{c-1} @ (W_A^64) + E_{c-1}; write every
// chunk's initial state to H0. 2 blocks (batch tiles), 64 sequential steps.
// ---------------------------------------------------------------------------
__global__ __launch_bounds__(512) void k_prefix(const unsigned short* __restrict__ WP,
                                                const float* __restrict__ E,
                                                float* __restrict__ H0) {
  __shared__ unsigned short h_sh[16 * 256];
  const int tid = threadIdx.x, lane = tid & 31, w = tid >> 5;
  const int bt = blockIdx.x, n0 = w * 16;
  v16bf wp[8];
#pragma unroll
  for (int kc = 0; kc < 8; ++kc) wp[kc] = bfrag(WP, 256, kc * 32, n0, lane);
  for (int i = tid; i < 16 * 256; i += 512) h_sh[i] = 0;
  v8f acc = zero8();
  __syncthreads();
  for (int c = 0; c < NCHUNK; ++c) {
    const size_t base = (size_t)(c * BT_TILES + bt) * 16 * 256;
    acc_to_gmem_f32(H0 + base, 256, n0, lane, acc);           // h_init for chunk c
    v8f a = acc_from_gmem_f32(E + base, 256, n0, lane);       // C-operand = E_c
#pragma unroll
    for (int kc = 0; kc < 8; ++kc)
      a = wmma_bf16(afrag(h_sh, 256, kc * 32, lane), wp[kc], a);
    __syncthreads();
    acc_to_lds_bf16(h_sh, 256, n0, lane, a);
    acc = a;
    __syncthreads();
  }
}

// ---------------------------------------------------------------------------
// Pass 2: re-run local scan seeded with H0[chunk]; waves 0-7 also compute
// y_t = h_t @ W_C + x_t @ W_D and store directly to d_out.
// ---------------------------------------------------------------------------
__global__ __launch_bounds__(512, 1) void k_output(
    const float* __restrict__ x, const unsigned short* __restrict__ WAs,
    const unsigned short* __restrict__ WBs, const unsigned short* __restrict__ WCs,
    const unsigned short* __restrict__ WDs, const float* __restrict__ H0,
    float* __restrict__ y) {
  __shared__ unsigned short h_sh[16 * 256];
  __shared__ unsigned short x_sh[16 * 128];
  const int tid = threadIdx.x, lane = tid & 31, w = tid >> 5;
  const int chunk = blockIdx.x, bt = blockIdx.y;
  const int b0 = bt * 16, t0 = chunk * L_CHUNK;
  const int n0 = w * 16;

  v16bf wa[8], wb[4], wc[8], wd[4];
#pragma unroll
  for (int kc = 0; kc < 8; ++kc)
    wa[kc] = ldfrag(WAs + (w * 8 + kc) * FRAG_U16, lane);
#pragma unroll
  for (int kc = 0; kc < 4; ++kc)
    wb[kc] = ldfrag(WBs + (w * 4 + kc) * FRAG_U16, lane);
  if (w < 8) {  // output tiles only span 128 columns
#pragma unroll
    for (int kc = 0; kc < 8; ++kc)
      wc[kc] = ldfrag(WCs + (w * 8 + kc) * FRAG_U16, lane);
#pragma unroll
    for (int kc = 0; kc < 4; ++kc)
      wd[kc] = ldfrag(WDs + (w * 4 + kc) * FRAG_U16, lane);
  }

  {  // publish initial state h_{t0-1} = H0[chunk]
    v8f h0 = acc_from_gmem_f32(H0 + (size_t)(chunk * BT_TILES + bt) * 16 * 256,
                               256, n0, lane);
    acc_to_lds_bf16(h_sh, 256, n0, lane, h0);
  }
  const int e0 = tid * 4, srow = e0 >> 7, scol = e0 & 127;
  const float* xbase = x + (size_t)(b0 + srow) * SEQ_LEN * D_INPUT + scol;
  __syncthreads();

  for (int t = 0; t < L_CHUNK; ++t) {
    const float* xp = xbase + (size_t)(t0 + t) * D_INPUT;
    float4 xv = *(const float4*)xp;
    if (t + 1 < L_CHUNK) __builtin_prefetch(xp + D_INPUT, 0, 0);
    uint2 pk;
    pk.x = (unsigned)f2bf(xv.x) | ((unsigned)f2bf(xv.y) << 16);
    pk.y = (unsigned)f2bf(xv.z) | ((unsigned)f2bf(xv.w) << 16);
    *(uint2*)&x_sh[srow * 128 + scol] = pk;
    __syncthreads();

    // h_t = h_{t-1} @ W_A + x_t @ W_B
    v8f h = zero8();
#pragma unroll
    for (int kc = 0; kc < 8; ++kc)
      h = wmma_bf16(afrag(h_sh, 256, kc * 32, lane), wa[kc], h);
#pragma unroll
    for (int kc = 0; kc < 4; ++kc)
      h = wmma_bf16(afrag(x_sh, 128, kc * 32, lane), wb[kc], h);
    __syncthreads();                       // h_sh reads done
    acc_to_lds_bf16(h_sh, 256, n0, lane, h);
    __syncthreads();                       // h_t visible

    if (w < 8) {  // y_t = h_t @ W_C + x_t @ W_D
      v8f yv = zero8();
#pragma unroll
      for (int kc = 0; kc < 8; ++kc)
        yv = wmma_bf16(afrag(h_sh, 256, kc * 32, lane), wc[kc], yv);
#pragma unroll
      for (int kc = 0; kc < 4; ++kc)
        yv = wmma_bf16(afrag(x_sh, 128, kc * 32, lane), wd[kc], yv);
      const int half = lane >> 4, col = n0 + (lane & 15);
#pragma unroll
      for (int r = 0; r < 8; ++r)
        y[((size_t)(b0 + half * 8 + r) * SEQ_LEN + (t0 + t)) * D_OUTPUT + col] =
            yv[r];
    }
    __syncthreads();                       // protect x_sh/h_sh for next step
  }
}

// ---------------------------------------------------------------------------
extern "C" void kernel_launch(void* const* d_in, const int* in_sizes, int n_in,
                              void* d_out, int out_size, void* d_ws, size_t ws_size,
                              hipStream_t stream) {
  const float* x   = (const float*)d_in[0];
  const float* A   = (const float*)d_in[1];
  const float* B   = (const float*)d_in[2];
  const float* C   = (const float*)d_in[3];
  const float* D   = (const float*)d_in[4];
  const float* ldt = (const float*)d_in[5];
  float* y = (float*)d_out;

  char* ws = (char*)d_ws;
  unsigned short* WA  = (unsigned short*)(ws);            // raw 256x256 bf16
  unsigned short* SQ0 = (unsigned short*)(ws + 131072);   // 256x256
  unsigned short* SQ1 = (unsigned short*)(ws + 262144);   // 256x256
  unsigned short* WAs = (unsigned short*)(ws + 393216);   // swz 16x8 frags
  unsigned short* WBs = (unsigned short*)(ws + 524288);   // swz 16x4 frags
  unsigned short* WCs = (unsigned short*)(ws + 589824);   // swz 8x8 frags
  unsigned short* WDs = (unsigned short*)(ws + 655360);   // swz 8x4 frags
  float* E  = (float*)(ws + 688128);                      // 64*32*256 f32
  float* H0 = (float*)(ws + 688128 + 2097152);            // 64*32*256 f32

  k_setup<<<64, 256, 0, stream>>>(A, B, C, D, ldt, WA, WAs, WBs, WCs, WDs);
  k_matsq<<<16, 512, 0, stream>>>(WA,  SQ0);   // W_A^2
  k_matsq<<<16, 512, 0, stream>>>(SQ0, SQ1);   // W_A^4
  k_matsq<<<16, 512, 0, stream>>>(SQ1, SQ0);   // W_A^8
  k_matsq<<<16, 512, 0, stream>>>(SQ0, SQ1);   // W_A^16
  k_matsq<<<16, 512, 0, stream>>>(SQ1, SQ0);   // W_A^32
  k_matsq<<<16, 512, 0, stream>>>(SQ0, SQ1);   // W_A^64
  k_endstate<<<dim3(NCHUNK, BT_TILES), 512, 0, stream>>>(x, WAs, WBs, E);
  k_prefix<<<BT_TILES, 512, 0, stream>>>(SQ1, E, H0);
  k_output<<<dim3(NCHUNK, BT_TILES), 512, 0, stream>>>(x, WAs, WBs, WCs, WDs, H0, y);
}
